// RBFKernelLayer_65481071408364
// MI455X (gfx1250) — matrix operand order repaired and checked by reading, not compiled, not run on hardware
//
#include <hip/hip_runtime.h>

// ---------- types ----------
typedef __bf16        bf16x16 __attribute__((ext_vector_type(16)));
typedef float         f32x8   __attribute__((ext_vector_type(8)));
typedef unsigned int  u32x4   __attribute__((ext_vector_type(4)));

union Frag {
    bf16x16 v;
    u32x4   q[2];
};

#define RBF_N     8192
#define RBF_D     128
#define RBF_GAMMA 0.01f
#define TILES     (RBF_N / 16)   // 512
#define WAVES_PER_BLOCK 8

// ---------- prep: fp32 x -> bf16 hi/lo split + row norms ----------
__device__ __forceinline__ unsigned short f32_to_bf16_rne(float f) {
    unsigned int u = __float_as_uint(f);
    unsigned int r = u + 0x7fffu + ((u >> 16) & 1u);   // round-to-nearest-even
    return (unsigned short)(r >> 16);
}

__global__ __launch_bounds__(RBF_D)
void rbf_prep_kernel(const float* __restrict__ x,
                     unsigned short* __restrict__ xhi,
                     unsigned short* __restrict__ xlo,
                     float* __restrict__ sq) {
    const int row = blockIdx.x;
    const int t   = threadIdx.x;
    const size_t idx = (size_t)row * RBF_D + t;

    float v = x[idx];

    unsigned short h = f32_to_bf16_rne(v);
    float hf = __uint_as_float(((unsigned int)h) << 16);
    float lo = v - hf;
    unsigned short l = f32_to_bf16_rne(lo);

    xhi[idx] = h;
    xlo[idx] = l;

    __shared__ float red[RBF_D];
    red[t] = v * v;
    __syncthreads();
    for (int s = RBF_D / 2; s > 0; s >>= 1) {
        if (t < s) red[t] += red[t + s];
        __syncthreads();
    }
    if (t == 0) sq[row] = red[0];
}

// ---------- fragment loaders (per ISA 16-bit WMMA VGPR layouts) ----------
// A (16x32, MxK): lane m = lane&15; VGPR0-3 hold K = 8*lhi + 0..7,
//                 VGPR4-7 hold K = 16 + 8*lhi + 0..7  -> two 16B loads, 32B apart
__device__ __forceinline__ Frag load_fragA(const unsigned short* __restrict__ base,
                                           int row, int koff) {
    Frag f;
    const u32x4* p = (const u32x4*)(base + (size_t)row * RBF_D + koff);
    f.q[0] = p[0];
    f.q[1] = p[2];   // +16 elements
    return f;
}
// B (32x16, KxN): lane n = lane&15; VGPR0-3 hold K = 8*lhi + 0..7,
//                 VGPR4-7 hold K = 16 + 8*lhi + 0..7  -> two 16B loads, 32B apart
__device__ __forceinline__ Frag load_fragB(const unsigned short* __restrict__ base,
                                           int row, int koff) {
    Frag f;
    const u32x4* p = (const u32x4*)(base + (size_t)row * RBF_D + koff);
    f.q[0] = p[0];
    f.q[1] = p[2];   // +16 elements
    return f;
}

// ---------- fused main kernel ----------
__global__ __launch_bounds__(32 * WAVES_PER_BLOCK)
void rbf_main_kernel(const unsigned short* __restrict__ xhi,
                     const unsigned short* __restrict__ xlo,
                     const float* __restrict__ sq,
                     const float* __restrict__ w,
                     float* __restrict__ out) {
    const int tid  = threadIdx.x;
    const int wave = tid >> 5;
    const int lane = tid & 31;
    const int lhi  = lane >> 4;       // half-wave select
    const int lmod = lane & 15;
    const int i0   = blockIdx.x * 16; // 16 output rows per workgroup

    // Resident A-tile fragments: x[i0:i0+16, 0:128], hi + lo split
    Frag ahi[4], alo[4];
#pragma unroll
    for (int c = 0; c < 4; ++c) {
        const int ka = 32 * c + 8 * lhi;
        ahi[c] = load_fragA(xhi, i0 + lmod, ka);
        alo[c] = load_fragA(xlo, i0 + lmod, ka);
    }

    // Row norms for the rows this lane's C-fragment covers (M = r + 8*lhi)
    float sqi[8];
#pragma unroll
    for (int r = 0; r < 8; ++r) sqi[r] = sq[i0 + r + 8 * lhi];

    float acc[8] = {0.f, 0.f, 0.f, 0.f, 0.f, 0.f, 0.f, 0.f};

    // Waves stride across the 512 j-tiles (uniform trip count: EXEC stays full)
    for (int t = wave; t < TILES; t += WAVES_PER_BLOCK) {
        const int j0 = t * 16;

        // prefetch next j-tile row for this lane
        const int tn = t + WAVES_PER_BLOCK;
        if (tn < TILES) {
            __builtin_prefetch(xhi + (size_t)(tn * 16 + lmod) * RBF_D, 0, 0);
            __builtin_prefetch(xlo + (size_t)(tn * 16 + lmod) * RBF_D, 0, 0);
        }

        Frag bhi[4], blo[4];
#pragma unroll
        for (int c = 0; c < 4; ++c) {
            const int kb = 32 * c + 8 * lhi;
            bhi[c] = load_fragB(xhi, j0 + lmod, kb);
            blo[c] = load_fragB(xlo, j0 + lmod, kb);
        }

        // Gram tile via bf16x3: hi*hi + hi*lo + lo*hi  (~fp32 accuracy)
        f32x8 g = {0.f, 0.f, 0.f, 0.f, 0.f, 0.f, 0.f, 0.f};
#pragma unroll
        for (int c = 0; c < 4; ++c) {
            g = __builtin_amdgcn_wmma_f32_16x16x32_bf16(
                    false, ahi[c].v, false, bhi[c].v, (short)0, g, false, false);
            g = __builtin_amdgcn_wmma_f32_16x16x32_bf16(
                    false, ahi[c].v, false, blo[c].v, (short)0, g, false, false);
            g = __builtin_amdgcn_wmma_f32_16x16x32_bf16(
                    false, alo[c].v, false, bhi[c].v, (short)0, g, false, false);
        }

        // Fused RBF epilogue: dist^2 -> exp -> * weight[j], accumulate row sums
        const float sqj = sq[j0 + lmod];
        const float wj  = w[j0 + lmod];
#pragma unroll
        for (int r = 0; r < 8; ++r) {
            float d = sqi[r] + sqj - 2.0f * g[r];
            d = fmaxf(d, 0.0f);
            acc[r] += __expf(-RBF_GAMMA * d) * wj;
        }
    }

    // Reduce over the 16 N-columns held by each half-wave (deterministic)
#pragma unroll
    for (int m = 1; m < 16; m <<= 1) {
#pragma unroll
        for (int r = 0; r < 8; ++r) acc[r] += __shfl_xor(acc[r], m, 16);
    }

    // Deterministic ordered cross-wave reduction in LDS
    __shared__ float part[WAVES_PER_BLOCK][16];
    if (lmod == 0) {
#pragma unroll
        for (int r = 0; r < 8; ++r) part[wave][r + 8 * lhi] = acc[r];
    }
    __syncthreads();

    if (tid < 16) {
        float s = 0.f;
#pragma unroll
        for (int wv = 0; wv < WAVES_PER_BLOCK; ++wv) s += part[wv][tid];
        out[i0 + tid] = s;
    }
}

// ---------- launch ----------
extern "C" void kernel_launch(void* const* d_in, const int* in_sizes, int n_in,
                              void* d_out, int out_size, void* d_ws, size_t ws_size,
                              hipStream_t stream) {
    const float* x      = (const float*)d_in[0];  // [N, D] fp32
    const float* weight = (const float*)d_in[1];  // [N] fp32
    float*       out    = (float*)d_out;          // [N] fp32

    char* ws = (char*)d_ws;
    unsigned short* xhi = (unsigned short*)ws;                                    // 2 MB
    unsigned short* xlo = (unsigned short*)(ws + (size_t)RBF_N * RBF_D * 2);      // 2 MB
    float*          sq  = (float*)        (ws + (size_t)RBF_N * RBF_D * 4);      // 32 KB

    rbf_prep_kernel<<<RBF_N, RBF_D, 0, stream>>>(x, xhi, xlo, sq);
    rbf_main_kernel<<<RBF_N / 16, 32 * WAVES_PER_BLOCK, 0, stream>>>(
        xhi, xlo, sq, weight, out);
}
